// YOLOv1_74792560492944
// MI455X (gfx1250) — compile-verified, matrix operand order checked by reading, not run on hardware
//
#include <hip/hip_runtime.h>
#include <math.h>

#define HEADC 512
#define FMPX 40
#define PADW (FMPX + 2)             // 42 (zero-padded spatial)
#define BATCH 8
#define PIX_PER_B (FMPX * FMPX)     // 1600
#define NPIX (BATCH * PIX_PER_B)    // 12800
#define PPB (PADW * PADW)           // 1764 padded pixels per image
#define NPPIX (BATCH * PPB)         // 14112
#define NCLS 20
#define CONF_T 0.001f
#define NMS_T 0.6f

typedef __attribute__((ext_vector_type(16))) _Float16 v16h;
typedef __attribute__((ext_vector_type(8)))  float    v8f;

union FragA { v16h v; uint4 q[2]; };
union Out8  { uint4 q; _Float16 h[8]; };

// ---------------------------------------------------------------------------
// Zero fill (uint4 granularity) — clears padded activation buffers so the
// conv kernels never need boundary checks.
// ---------------------------------------------------------------------------
__global__ void k_zero(uint4* __restrict__ p, long long n16) {
  long long g = (long long)blockIdx.x * blockDim.x + threadIdx.x;
  if (g < n16) p[g] = make_uint4(0u, 0u, 0u, 0u);
}

// ---------------------------------------------------------------------------
// x: (B, 512, 40, 40) fp32 NCHW -> padded NHWC fp16 [b][y+1][x+1][c]
// ---------------------------------------------------------------------------
__global__ void k_cvt_x(const float* __restrict__ x, _Float16* __restrict__ out) {
  int g = blockIdx.x * blockDim.x + threadIdx.x;
  if (g >= NPIX * HEADC) return;
  int p = g >> 9;              // /512
  int c = g & 511;
  int b = p / PIX_PER_B;
  int rem = p - b * PIX_PER_B;
  int y = rem / FMPX;
  int xx = rem - y * FMPX;
  size_t po = (size_t)(b * PPB + (y + 1) * PADW + (xx + 1)) * HEADC + c;
  out[po] = (_Float16)x[(size_t)(b * HEADC + c) * PIX_PER_B + rem];
}

// ---------------------------------------------------------------------------
// w: (O=512, I=512, 3, 3) fp32 OIHW -> wt[tap=ky*3+kx][O][I] fp16
// ---------------------------------------------------------------------------
__global__ void k_cvt_w(const float* __restrict__ w, _Float16* __restrict__ wt) {
  int g = blockIdx.x * blockDim.x + threadIdx.x;
  if (g >= 9 * HEADC * HEADC) return;
  int tap = g / (HEADC * HEADC);
  int r = g - tap * HEADC * HEADC;
  int o = r >> 9;
  int i = r & 511;
  wt[g] = (_Float16)w[(size_t)(o * HEADC + i) * 9 + tap];
}

// ---------------------------------------------------------------------------
// 3x3 conv as implicit GEMM with WMMA f32 <- f16*f16 on padded NHWC buffers.
// M = Cout = 512, N = 12800 output pixels, K = 512 per tap, 9 taps.
// Per wave: 32(M) x 64(N) tile = 2x4 accumulators; 8 WMMA per 12 b128 loads.
// No boundary checks anywhere (inputs are zero-padded).
// ---------------------------------------------------------------------------
__global__ __launch_bounds__(128) void k_conv3(const _Float16* __restrict__ in,
                                               const _Float16* __restrict__ wt,
                                               const float* __restrict__ bias,
                                               _Float16* __restrict__ out) {
  const int wave  = blockIdx.x * 4 + (threadIdx.x >> 5);
  const int lane  = threadIdx.x & 31;
  const int lidx  = lane & 15;
  const int half  = lane >> 4;      // 0: K 0..7/16..23, 1: K 8..15/24..31
  const int mTile = (wave & 15) * 32;
  const int nTile = (wave >> 4) * 64;

  // Padded center index for each of the 4 N sub-tiles' pixels.
  int ctr[4];
#pragma unroll
  for (int j = 0; j < 4; ++j) {
    int p = nTile + j * 16 + lidx;
    int b = p / PIX_PER_B;
    int rem = p - b * PIX_PER_B;
    int y = rem / FMPX;
    int xx = rem - y * FMPX;
    ctr[j] = b * PPB + (y + 1) * PADW + (xx + 1);
  }

  v8f acc0[4] = {{}, {}, {}, {}};   // mh = 0
  v8f acc1[4] = {{}, {}, {}, {}};   // mh = 1
  const int kLaneOff = 8 * half;

  for (int tap = 0; tap < 9; ++tap) {
    const int dy = tap / 3 - 1;
    const int dx = tap % 3 - 1;
    const int tapOff = (dy * PADW + dx) * HEADC;
    const _Float16* pa0 = wt + ((size_t)tap * HEADC + (mTile + lidx)) * HEADC + kLaneOff;
    const _Float16* pa1 = wt + ((size_t)tap * HEADC + (mTile + 16 + lidx)) * HEADC + kLaneOff;
    const _Float16* pb[4];
#pragma unroll
    for (int j = 0; j < 4; ++j)
      pb[j] = in + (size_t)ctr[j] * HEADC + tapOff + kLaneOff;

    for (int kc = 0; kc < HEADC; kc += 32) {
      FragA a0, a1;
      a0.q[0] = *reinterpret_cast<const uint4*>(pa0 + kc);
      a0.q[1] = *reinterpret_cast<const uint4*>(pa0 + kc + 16);
      a1.q[0] = *reinterpret_cast<const uint4*>(pa1 + kc);
      a1.q[1] = *reinterpret_cast<const uint4*>(pa1 + kc + 16);
      FragA fb[4];
#pragma unroll
      for (int j = 0; j < 4; ++j) {
        fb[j].q[0] = *reinterpret_cast<const uint4*>(pb[j] + kc);
        fb[j].q[1] = *reinterpret_cast<const uint4*>(pb[j] + kc + 16);
      }
#pragma unroll
      for (int j = 0; j < 4; ++j) {
        acc0[j] = __builtin_amdgcn_wmma_f32_16x16x32_f16(false, a0.v, false, fb[j].v,
                                                         (short)0, acc0[j], false, false);
        acc1[j] = __builtin_amdgcn_wmma_f32_16x16x32_f16(false, a1.v, false, fb[j].v,
                                                         (short)0, acc1[j], false, false);
      }
    }
  }

  // Epilogue: bias + leaky ReLU, fp16 store into padded interior.
  auto store_tile = [&](v8f acc, int chBase, int pctr) {
    Out8 st;
#pragma unroll
    for (int r = 0; r < 8; ++r) {
      float v = acc[r] + bias[chBase + r];
      v = v > 0.0f ? v : 0.1f * v;
      st.h[r] = (_Float16)v;
    }
    *reinterpret_cast<uint4*>(out + (size_t)pctr * HEADC + chBase) = st.q;
  };
#pragma unroll
  for (int j = 0; j < 4; ++j) {
    store_tile(acc0[j], mTile + 8 * half,      ctr[j]);
    store_tile(acc1[j], mTile + 16 + 8 * half, ctr[j]);
  }
}

// ---------------------------------------------------------------------------
// 1x1 conv heads: obj(1 from rf), cls(20 from cf), reg(4 from rf).
// One thread per (pixel, output) pair; activations read from padded NHWC.
// ---------------------------------------------------------------------------
__global__ void k_heads(const _Float16* __restrict__ cf, const _Float16* __restrict__ rf,
                        const float* __restrict__ ow, const float* __restrict__ ob,
                        const float* __restrict__ cw, const float* __restrict__ cb,
                        const float* __restrict__ rw, const float* __restrict__ rb,
                        float* __restrict__ obj, float* __restrict__ cls,
                        float* __restrict__ reg) {
  int t = blockIdx.x * blockDim.x + threadIdx.x;
  if (t >= NPIX * 25) return;
  int p = t / 25;
  int o = t - p * 25;
  int b = p / PIX_PER_B;
  int rem = p - b * PIX_PER_B;
  int y = rem / FMPX;
  int xx = rem - y * FMPX;
  size_t pp = (size_t)(b * PPB + (y + 1) * PADW + (xx + 1));

  const _Float16* act;
  const float* w;
  float bi;
  float* dst;
  if (o == 0)         { act = rf; w = ow;                  bi = ob[0]; dst = obj + p; }
  else if (o <= NCLS) { int j = o - 1;  act = cf; w = cw + j * HEADC; bi = cb[j]; dst = cls + p * NCLS + j; }
  else                { int j = o - 21; act = rf; w = rw + j * HEADC; bi = rb[j]; dst = reg + p * 4 + j; }
  const _Float16* a = act + pp * HEADC;
  float s = bi;
  for (int c = 0; c < HEADC; ++c) s += w[c] * (float)a[c];
  *dst = s;
}

// ---------------------------------------------------------------------------
// Decode: softmax/argmax over classes, sigmoid(obj), box decode + clip.
// ---------------------------------------------------------------------------
__global__ void k_decode(const float* __restrict__ obj, const float* __restrict__ cls,
                         const float* __restrict__ reg, float* __restrict__ bboxes,
                         float* __restrict__ best, int* __restrict__ clsInd) {
  int p = blockIdx.x * blockDim.x + threadIdx.x;
  if (p >= NPIX) return;
  float m = -1e30f; int bj = 0;
  for (int j = 0; j < NCLS; ++j) {
    float v = cls[p * NCLS + j];
    if (v > m) { m = v; bj = j; }
  }
  float se = 0.0f;
  for (int j = 0; j < NCLS; ++j) se += __expf(cls[p * NCLS + j] - m);
  float so = 1.0f / (1.0f + __expf(-obj[p]));
  float bestv = so / se;     // so * exp(0)/se at the argmax

  float tx = 1.0f / (1.0f + __expf(-reg[p * 4 + 0]));
  float ty = 1.0f / (1.0f + __expf(-reg[p * 4 + 1]));
  float bw = __expf(reg[p * 4 + 2]);
  float bh = __expf(reg[p * 4 + 3]);
  int n  = p % PIX_PER_B;
  int gy = n / FMPX;
  int gx = n - gy * FMPX;
  float cx = tx + (float)gx;
  float cy = ty + (float)gy;
  const float sc = 32.0f / 1280.0f;
  float X1 = fminf(fmaxf((cx - 0.5f * bw) * sc, 0.0f), 1.0f);
  float Y1 = fminf(fmaxf((cy - 0.5f * bh) * sc, 0.0f), 1.0f);
  float X2 = fminf(fmaxf((cx + 0.5f * bw) * sc, 0.0f), 1.0f);
  float Y2 = fminf(fmaxf((cy + 0.5f * bh) * sc, 0.0f), 1.0f);
  bboxes[p * 4 + 0] = X1;
  bboxes[p * 4 + 1] = Y1;
  bboxes[p * 4 + 2] = X2;
  bboxes[p * 4 + 3] = Y2;
  best[p] = bestv;
  clsInd[p] = bj;
}

// ---------------------------------------------------------------------------
// Greedy per-class NMS, one block per batch image, everything staged in LDS.
// ---------------------------------------------------------------------------
__global__ __launch_bounds__(256) void k_nms(const float* __restrict__ bboxes,
                                             const float* __restrict__ best,
                                             const int* __restrict__ clsInd,
                                             float* __restrict__ keepOut) {
  __shared__ float sx1[PIX_PER_B], sy1[PIX_PER_B], sx2[PIX_PER_B], sy2[PIX_PER_B];
  __shared__ float sar[PIX_PER_B], ssc[PIX_PER_B];
  __shared__ int   sord[PIX_PER_B], scl[PIX_PER_B];
  __shared__ unsigned char skp[PIX_PER_B];

  const int b = blockIdx.x;
  const int tid = threadIdx.x;
  const float* bb = bboxes + (size_t)b * PIX_PER_B * 4;
  const float* bs = best + (size_t)b * PIX_PER_B;
  const int*   ci = clsInd + (size_t)b * PIX_PER_B;

  for (int i = tid; i < PIX_PER_B; i += blockDim.x) {
    float X1 = bb[i * 4 + 0], Y1 = bb[i * 4 + 1];
    float X2 = bb[i * 4 + 2], Y2 = bb[i * 4 + 3];
    sx1[i] = X1; sy1[i] = Y1; sx2[i] = X2; sy2[i] = Y2;
    sar[i] = (X2 - X1) * (Y2 - Y1);
    float s = bs[i];
    bool conf = (s >= CONF_T);
    ssc[i] = conf ? s : -INFINITY;
    scl[i] = ci[i];
    skp[i] = conf ? 1 : 0;
  }
  __syncthreads();

  // Stable descending rank (ties broken by ascending index, like argsort).
  for (int i = tid; i < PIX_PER_B; i += blockDim.x) {
    float si = ssc[i];
    int cnt = 0;
    for (int j = 0; j < PIX_PER_B; ++j) {
      float sj = ssc[j];
      if (sj > si || (sj == si && j < i)) cnt++;
    }
    sord[cnt] = i;
  }
  __syncthreads();

  for (int pos = 0; pos < PIX_PER_B; ++pos) {
    int idx = sord[pos];
    if (skp[idx]) {
      float ax1 = sx1[idx], ay1 = sy1[idx], ax2 = sx2[idx], ay2 = sy2[idx];
      float aa = sar[idx];
      int ac = scl[idx];
      for (int j = pos + 1 + tid; j < PIX_PER_B; j += blockDim.x) {
        int jd = sord[j];
        if (skp[jd] && scl[jd] == ac) {
          float xx1 = fmaxf(ax1, sx1[jd]);
          float yy1 = fmaxf(ay1, sy1[jd]);
          float xx2 = fminf(ax2, sx2[jd]);
          float yy2 = fminf(ay2, sy2[jd]);
          float w = fmaxf(1e-28f, xx2 - xx1);
          float h = fmaxf(1e-28f, yy2 - yy1);
          float inter = w * h;
          float iou = inter / (aa + sar[jd] - inter + 1e-14f);
          if (iou > NMS_T) skp[jd] = 0;
        }
      }
    }
    __syncthreads();
  }

  for (int i = tid; i < PIX_PER_B; i += blockDim.x)
    keepOut[(size_t)b * PIX_PER_B + i] = skp[i] ? 1.0f : 0.0f;
}

// ---------------------------------------------------------------------------
extern "C" void kernel_launch(void* const* d_in, const int* in_sizes, int n_in,
                              void* d_out, int out_size, void* d_ws, size_t ws_size,
                              hipStream_t stream) {
  (void)in_sizes; (void)n_in; (void)out_size; (void)ws_size;

  const float* x      = (const float*)d_in[0];
  const float* cls_w  = (const float*)d_in[1];
  const float* cls_b  = (const float*)d_in[2];
  const float* reg_w  = (const float*)d_in[3];
  const float* reg_b  = (const float*)d_in[4];
  const float* obj_w  = (const float*)d_in[5];
  const float* obj_b  = (const float*)d_in[6];
  const float* clsp_w = (const float*)d_in[7];
  const float* clsp_b = (const float*)d_in[8];
  const float* regp_w = (const float*)d_in[9];
  const float* regp_b = (const float*)d_in[10];

  char* ws = (char*)d_ws;
  const size_t actBytes = (size_t)NPPIX * HEADC * sizeof(_Float16);   // ~14.5 MB (padded)
  _Float16* X  = (_Float16*)(ws + 0 * actBytes);
  _Float16* T1 = (_Float16*)(ws + 1 * actBytes);
  _Float16* T2 = (_Float16*)(ws + 2 * actBytes);   // cf
  _Float16* T3 = (_Float16*)(ws + 3 * actBytes);   // rf
  const size_t wBytes = (size_t)9 * HEADC * HEADC * sizeof(_Float16); // ~4.7 MB
  _Float16* W[6];
  for (int i = 0; i < 6; ++i) W[i] = (_Float16*)(ws + 4 * actBytes + (size_t)i * wBytes);
  float* objA = (float*)(ws + 4 * actBytes + 6 * wBytes);
  float* clsA = objA + NPIX;
  float* regA = clsA + (size_t)NPIX * NCLS;

  // Zero padded activation buffers (borders must be 0; interiors get overwritten).
  {
    long long n16 = (long long)(4 * actBytes) / 16;
    int blocks = (int)((n16 + 255) / 256);
    k_zero<<<blocks, 256, 0, stream>>>((uint4*)ws, n16);
  }

  // Layout conversions.
  k_cvt_x<<<(NPIX * HEADC + 255) / 256, 256, 0, stream>>>(x, X);
  const size_t ls = (size_t)HEADC * HEADC * 9;
  const int wg = (int)((ls + 255) / 256);
  k_cvt_w<<<wg, 256, 0, stream>>>(cls_w,          W[0]);
  k_cvt_w<<<wg, 256, 0, stream>>>(cls_w + ls,     W[1]);
  k_cvt_w<<<wg, 256, 0, stream>>>(reg_w,          W[2]);
  k_cvt_w<<<wg, 256, 0, stream>>>(reg_w + 1 * ls, W[3]);
  k_cvt_w<<<wg, 256, 0, stream>>>(reg_w + 2 * ls, W[4]);
  k_cvt_w<<<wg, 256, 0, stream>>>(reg_w + 3 * ls, W[5]);

  // 6 WMMA convs: 16 x 200 = 3200 waves, 4 waves/block -> 800 blocks of 128.
  const dim3 cg(800), cb(128);
  k_conv3<<<cg, cb, 0, stream>>>(X,  W[0], cls_b,             T1);
  k_conv3<<<cg, cb, 0, stream>>>(T1, W[1], cls_b + HEADC,     T2);  // cf
  k_conv3<<<cg, cb, 0, stream>>>(X,  W[2], reg_b,             T1);
  k_conv3<<<cg, cb, 0, stream>>>(T1, W[3], reg_b + 1 * HEADC, T3);
  k_conv3<<<cg, cb, 0, stream>>>(T3, W[4], reg_b + 2 * HEADC, T1);
  k_conv3<<<cg, cb, 0, stream>>>(T1, W[5], reg_b + 3 * HEADC, T3);  // rf

  float* out    = (float*)d_out;
  float* bboxes = out;                 // (B,N,4) -> 51200 floats
  float* bestO  = out + NPIX * 4;      // (B,N)
  int*   clsO   = (int*)(out + NPIX * 5);
  float* keepO  = out + NPIX * 6;

  k_heads<<<(NPIX * 25 + 255) / 256, 256, 0, stream>>>(T2, T3, obj_w, obj_b,
                                                       clsp_w, clsp_b, regp_w, regp_b,
                                                       objA, clsA, regA);
  k_decode<<<(NPIX + 255) / 256, 256, 0, stream>>>(objA, clsA, regA, bboxes, bestO, clsO);
  k_nms<<<BATCH, 256, 0, stream>>>(bboxes, bestO, clsO, keepO);
}
